// WolfPQRotate_78520592106005
// MI455X (gfx1250) — compile-verified, matrix operand order checked by reading, not run on hardware
//
#include <hip/hip_runtime.h>
#include <math.h>

// ---------------------------------------------------------------------------
// WolfPQ encoder forward for MI455X (gfx1250, wave32, WMMA).
//   out  = codebook gather of argmax_k( relu(MLP(x @ R^T)) + gumbel(u) )
//   s    = one-hot of that argmax (exact forward value of the ST estimator)
// GEMMs run in f16 with f32 accumulation via v_wmma_f32_16x16x32_f16.
// Pipelined: next k-tile prefetched into registers while WMMAs run.
// u loads / s stores are non-temporal so h (100MB) + W2 slice stay in L2.
// ---------------------------------------------------------------------------

typedef __attribute__((ext_vector_type(16))) _Float16 v16h;
typedef __attribute__((ext_vector_type(8)))  _Float16 v8h;
typedef __attribute__((ext_vector_type(4)))  _Float16 v4h;
typedef __attribute__((ext_vector_type(8)))  float    v8f;
typedef __attribute__((ext_vector_type(4)))  float    v4f;

#define B_ROWS 8192
#define DIM    768
#define M_GRP  48
#define K_CB   256
#define H_DIM  6144
#define NKK    12288   // M_GRP * K_CB
#define DSUB   16

// ------------------------------- f32 -> f16 --------------------------------
__global__ __launch_bounds__(256) void cvt_f32_f16(const float* __restrict__ src,
                                                   _Float16* __restrict__ dst) {
  size_t i = ((size_t)blockIdx.x * 256 + threadIdx.x) * 4;
  v4f v = *(const v4f*)(src + i);
  v4h o;
  o[0] = (_Float16)v[0]; o[1] = (_Float16)v[1];
  o[2] = (_Float16)v[2]; o[3] = (_Float16)v[3];
  *(v4h*)(dst + i) = o;
}

// -------------------- generic 128x128 WMMA GEMM: C = act(A @ Bw^T + bias) ---
// A: [Mrows, K] f16 row-major.  Bw: [N, K] f16 row-major.  C: [Mrows, N] f16.
// ACT: 0 = identity (no bias), 1 = tanh(acc + bias[col]).
template <int ACT>
__global__ __launch_bounds__(256) void gemm_wmma_128x128(
    const _Float16* __restrict__ A, const _Float16* __restrict__ Bw,
    const float* __restrict__ bias, _Float16* __restrict__ C,
    int N, int K) {
  constexpr int LDP = 40;  // half-element pitch: 80B rows -> conflict-free frags
  __shared__ _Float16 sA[128 * LDP];
  __shared__ _Float16 sB[128 * LDP];

  const int t    = threadIdx.x;
  const int wave = t >> 5;
  const int lane = t & 31;
  const int wr   = wave >> 2;        // 0..1  (64-row wave tile)
  const int wc   = wave & 3;         // 0..3  (32-col wave tile)
  const int r    = lane & 15;
  const int hh   = lane >> 4;
  const int row0 = blockIdx.x * 128;
  const int col0 = blockIdx.y * 128;

  v8f acc[4][2];
#pragma unroll
  for (int mt = 0; mt < 4; ++mt)
#pragma unroll
    for (int nt = 0; nt < 2; ++nt)
#pragma unroll
      for (int v = 0; v < 8; ++v) acc[mt][nt][v] = 0.0f;

  // staging-chunk coordinates (2 chunks of A and B per thread)
  const int rr0 = t >> 2, rr1 = (t + 256) >> 2;
  const int cc0 = (t & 3) << 3;

  // prologue: prefetch k-tile 0 into registers
  v8h pa0, pa1, pb0, pb1;
  pa0 = *(const v8h*)&A[(size_t)(row0 + rr0) * K + cc0];
  pa1 = *(const v8h*)&A[(size_t)(row0 + rr1) * K + cc0];
  pb0 = *(const v8h*)&Bw[(size_t)(col0 + rr0) * K + cc0];
  pb1 = *(const v8h*)&Bw[(size_t)(col0 + rr1) * K + cc0];

  for (int k0 = 0; k0 < K; k0 += 32) {
    // commit prefetched tile to LDS
    *(v8h*)&sA[rr0 * LDP + cc0] = pa0;
    *(v8h*)&sA[rr1 * LDP + cc0] = pa1;
    *(v8h*)&sB[rr0 * LDP + cc0] = pb0;
    *(v8h*)&sB[rr1 * LDP + cc0] = pb1;
    __syncthreads();

    // issue next tile's global loads (overlap with WMMA below)
    if (k0 + 32 < K) {
      int kn = k0 + 32 + cc0;
      pa0 = *(const v8h*)&A[(size_t)(row0 + rr0) * K + kn];
      pa1 = *(const v8h*)&A[(size_t)(row0 + rr1) * K + kn];
      pb0 = *(const v8h*)&Bw[(size_t)(col0 + rr0) * K + kn];
      pb1 = *(const v8h*)&Bw[(size_t)(col0 + rr1) * K + kn];
    }

    v16h af[4], bf[2];
#pragma unroll
    for (int mt = 0; mt < 4; ++mt) {
      const _Float16* p = &sA[(wr * 64 + mt * 16 + r) * LDP + hh * 8];
      v8h lo = *(const v8h*)p;
      v8h hi = *(const v8h*)(p + 16);
      af[mt] = __builtin_shufflevector(lo, hi, 0, 1, 2, 3, 4, 5, 6, 7,
                                               8, 9, 10, 11, 12, 13, 14, 15);
    }
#pragma unroll
    for (int nt = 0; nt < 2; ++nt) {
      const _Float16* p = &sB[(wc * 32 + nt * 16 + r) * LDP + hh * 8];
      v8h lo = *(const v8h*)p;
      v8h hi = *(const v8h*)(p + 16);
      bf[nt] = __builtin_shufflevector(lo, hi, 0, 1, 2, 3, 4, 5, 6, 7,
                                               8, 9, 10, 11, 12, 13, 14, 15);
    }
#pragma unroll
    for (int mt = 0; mt < 4; ++mt)
#pragma unroll
      for (int nt = 0; nt < 2; ++nt)
        acc[mt][nt] = __builtin_amdgcn_wmma_f32_16x16x32_f16(
            false, af[mt], false, bf[nt], (short)0, acc[mt][nt], false, false);
    __syncthreads();
  }

  // epilogue: C/D layout -> (m_local = v + 8*hh, n_local = r)
#pragma unroll
  for (int mt = 0; mt < 4; ++mt)
#pragma unroll
    for (int nt = 0; nt < 2; ++nt) {
      int gcol = col0 + wc * 32 + nt * 16 + r;
#pragma unroll
      for (int v = 0; v < 8; ++v) {
        int grow = row0 + wr * 64 + mt * 16 + v + 8 * hh;
        float val = acc[mt][nt][v];
        if (ACT == 1) val = tanhf(val + bias[gcol]);
        C[(size_t)grow * N + gcol] = (_Float16)val;
      }
    }
}

// ----- GEMM3 fused: a = relu(h @ W2^T + b2); z = a + gumbel(u); argmax ------
// grid = (B_ROWS/64, M_GRP). Block tile: 64 rows x 256 cols (one m-group).
__global__ __launch_bounds__(256) void gemm3_gumbel_argmax(
    const _Float16* __restrict__ Hh, const _Float16* __restrict__ W2h,
    const float* __restrict__ b2, const float* __restrict__ u,
    const float* __restrict__ codebook, float* __restrict__ out,
    float* __restrict__ s, int K) {
  constexpr int LDP = 40;
  constexpr int CPITCH = 260;  // f32 pitch for epilogue spill
  __shared__ __align__(16) unsigned char smem[64 * CPITCH * 4];  // 66560 B
  _Float16* sA = (_Float16*)smem;              // 64  x 40 halves (5120 B)
  _Float16* sB = sA + 64 * LDP;                // 256 x 40 halves (20480 B)
  float*    smc = (float*)smem;                // 64 x 260 f32 (epilogue alias)
  __shared__ float redv[64][4];
  __shared__ int   redi[64][4];
  __shared__ int   rowidx[64];

  const int t    = threadIdx.x;
  const int wave = t >> 5;
  const int lane = t & 31;
  const int wr   = wave >> 2;      // 0..1 (32-row wave tile)
  const int wc   = wave & 3;       // 0..3 (64-col wave tile)
  const int r    = lane & 15;
  const int hh   = lane >> 4;
  const int row0 = blockIdx.x * 64;
  const int m    = blockIdx.y;
  const int n0   = m * K_CB;

  v8f acc[2][4];
#pragma unroll
  for (int mt = 0; mt < 2; ++mt)
#pragma unroll
    for (int nt = 0; nt < 4; ++nt)
#pragma unroll
      for (int v = 0; v < 8; ++v) acc[mt][nt][v] = 0.0f;

  // staging-chunk coordinates: 1 A chunk + 4 B chunks per thread
  const int arr = t >> 2;
  const int cc0 = (t & 3) << 3;

  v8h pa, pb[4];
  pa = *(const v8h*)&Hh[(size_t)(row0 + arr) * K + cc0];
#pragma unroll
  for (int sg = 0; sg < 4; ++sg) {
    int rr = (t + sg * 256) >> 2;
    pb[sg] = *(const v8h*)&W2h[(size_t)(n0 + rr) * K + cc0];
  }

  for (int k0 = 0; k0 < K; k0 += 32) {
    // commit prefetched tile to LDS
    *(v8h*)&sA[arr * LDP + cc0] = pa;
#pragma unroll
    for (int sg = 0; sg < 4; ++sg) {
      int rr = (t + sg * 256) >> 2;
      *(v8h*)&sB[rr * LDP + cc0] = pb[sg];
    }
    __syncthreads();

    // issue next tile's global loads (overlap with WMMA below)
    if (k0 + 32 < K) {
      int kn = k0 + 32 + cc0;
      pa = *(const v8h*)&Hh[(size_t)(row0 + arr) * K + kn];
#pragma unroll
      for (int sg = 0; sg < 4; ++sg) {
        int rr = (t + sg * 256) >> 2;
        pb[sg] = *(const v8h*)&W2h[(size_t)(n0 + rr) * K + kn];
      }
    }

    v16h af[2], bf[4];
#pragma unroll
    for (int mt = 0; mt < 2; ++mt) {
      const _Float16* p = &sA[(wr * 32 + mt * 16 + r) * LDP + hh * 8];
      v8h lo = *(const v8h*)p;
      v8h hi = *(const v8h*)(p + 16);
      af[mt] = __builtin_shufflevector(lo, hi, 0, 1, 2, 3, 4, 5, 6, 7,
                                               8, 9, 10, 11, 12, 13, 14, 15);
    }
#pragma unroll
    for (int nt = 0; nt < 4; ++nt) {
      const _Float16* p = &sB[(wc * 64 + nt * 16 + r) * LDP + hh * 8];
      v8h lo = *(const v8h*)p;
      v8h hi = *(const v8h*)(p + 16);
      bf[nt] = __builtin_shufflevector(lo, hi, 0, 1, 2, 3, 4, 5, 6, 7,
                                               8, 9, 10, 11, 12, 13, 14, 15);
    }
#pragma unroll
    for (int mt = 0; mt < 2; ++mt)
#pragma unroll
      for (int nt = 0; nt < 4; ++nt)
        acc[mt][nt] = __builtin_amdgcn_wmma_f32_16x16x32_f16(
            false, af[mt], false, bf[nt], (short)0, acc[mt][nt], false, false);
    __syncthreads();
  }

  // ---- epilogue: spill accumulators to LDS (reuses staging memory) ----
#pragma unroll
  for (int mt = 0; mt < 2; ++mt)
#pragma unroll
    for (int nt = 0; nt < 4; ++nt) {
      int col = wc * 64 + nt * 16 + r;
#pragma unroll
      for (int v = 0; v < 8; ++v) {
        int row = wr * 32 + mt * 16 + v + 8 * hh;
        smc[row * CPITCH + col] = acc[mt][nt][v];
      }
    }
  __syncthreads();

  // ---- bias + relu + gumbel + partial argmax (4 threads per row) ----
  {
    int row = t & 63;
    int q   = t >> 6;
    int rowg = row0 + row;
    const float* up = u + ((size_t)rowg * M_GRP + m) * K_CB + q * 64;
    float bmax = -3.402823466e38f;
    int bidx = q * 64;
    for (int c = 0; c < 64; ++c) {
      int cc = q * 64 + c;
      float a = smc[row * CPITCH + cc] + b2[n0 + cc];
      a = a > 0.0f ? a : 0.0f;
      float uv = __builtin_nontemporal_load(up + c);  // NT: keep h/W2 in L2
      float g = -logf(-logf(uv));
      float z = a + g;
      if (z > bmax) { bmax = z; bidx = cc; }
    }
    redv[row][q] = bmax;
    redi[row][q] = bidx;
  }
  __syncthreads();

  // ---- final reduce + codebook gather (one thread per row) ----
  if (t < 64) {
    int row = t;
    float bmax = redv[row][0];
    int bidx = redi[row][0];
#pragma unroll
    for (int q = 1; q < 4; ++q) {
      if (redv[row][q] > bmax) { bmax = redv[row][q]; bidx = redi[row][q]; }
    }
    rowidx[row] = bidx;
    const v4f* cb = (const v4f*)(codebook + ((size_t)m * K_CB + bidx) * DSUB);
    v4f* op = (v4f*)(out + (size_t)(row0 + row) * DIM + m * DSUB);
#pragma unroll
    for (int d = 0; d < 4; ++d) op[d] = cb[d];
  }
  __syncthreads();

  // ---- one-hot write of s (non-temporal: 402MB stream, bypass L2 lines) ----
  {
    int row = t & 63;
    int q   = t >> 6;
    int idx = rowidx[row];
    v4f* sp = (v4f*)(s + ((size_t)(row0 + row) * M_GRP + m) * K_CB + q * 64);
    for (int c = 0; c < 64; c += 4) {
      int cc = q * 64 + c;
      v4f vv;
      vv[0] = (cc + 0 == idx) ? 1.0f : 0.0f;
      vv[1] = (cc + 1 == idx) ? 1.0f : 0.0f;
      vv[2] = (cc + 2 == idx) ? 1.0f : 0.0f;
      vv[3] = (cc + 3 == idx) ? 1.0f : 0.0f;
      __builtin_nontemporal_store(vv, sp + (c >> 2));
    }
  }
}

// ---------------------------------------------------------------------------
extern "C" void kernel_launch(void* const* d_in, const int* in_sizes, int n_in,
                              void* d_out, int out_size, void* d_ws,
                              size_t ws_size, hipStream_t stream) {
  const float* x        = (const float*)d_in[0];  // [B, DIM]
  const float* u        = (const float*)d_in[1];  // [B, M, K]
  const float* R        = (const float*)d_in[2];  // [DIM, DIM]
  const float* W1       = (const float*)d_in[3];  // [H, DIM]
  const float* b1       = (const float*)d_in[4];  // [H]
  const float* W2       = (const float*)d_in[5];  // [M*K, H]
  const float* b2       = (const float*)d_in[6];  // [M*K]
  const float* codebook = (const float*)d_in[7];  // [M, K, DSUB]

  float* out = (float*)d_out;                       // [B, DIM]
  float* s   = out + (size_t)B_ROWS * DIM;          // [B, M, K]

  // workspace carve (all f16 buffers, 16B aligned sizes)
  char* w = (char*)d_ws;
  _Float16* x16    = (_Float16*)w;  w += (size_t)B_ROWS * DIM * 2;
  _Float16* R16    = (_Float16*)w;  w += (size_t)DIM * DIM * 2;
  _Float16* W1_16  = (_Float16*)w;  w += (size_t)H_DIM * DIM * 2;
  _Float16* W2_16  = (_Float16*)w;  w += (size_t)NKK * H_DIM * 2;
  _Float16* xrot16 = (_Float16*)w;  w += (size_t)B_ROWS * DIM * 2;
  _Float16* h16    = (_Float16*)w;  w += (size_t)B_ROWS * H_DIM * 2;
  (void)ws_size; (void)n_in; (void)in_sizes; (void)out_size;

  // f32 -> f16 conversions (all element counts divisible by 1024)
  cvt_f32_f16<<<(B_ROWS * DIM) / 1024, 256, 0, stream>>>(x, x16);
  cvt_f32_f16<<<(DIM * DIM) / 1024, 256, 0, stream>>>(R, R16);
  cvt_f32_f16<<<(H_DIM * DIM) / 1024, 256, 0, stream>>>(W1, W1_16);
  cvt_f32_f16<<<((size_t)NKK * H_DIM) / 1024, 256, 0, stream>>>(W2, W2_16);

  // GEMM1: x_rot = x @ R^T
  gemm_wmma_128x128<0><<<dim3(B_ROWS / 128, DIM / 128), 256, 0, stream>>>(
      x16, R16, nullptr, xrot16, DIM, DIM);
  // GEMM2: h = tanh(x_rot @ W1^T + b1)
  gemm_wmma_128x128<1><<<dim3(B_ROWS / 128, H_DIM / 128), 256, 0, stream>>>(
      xrot16, W1_16, b1, h16, H_DIM, DIM);
  // GEMM3 fused: relu + gumbel + argmax + one-hot + codebook gather
  gemm3_gumbel_argmax<<<dim3(B_ROWS / 64, M_GRP), 256, 0, stream>>>(
      h16, W2_16, b2, u, codebook, out, s, H_DIM);
}
